// MHA_85392539779166
// MI455X (gfx1250) — matrix operand order repaired and checked by reading, not compiled, not run on hardware
//
#include <hip/hip_runtime.h>
#include <hip/hip_bf16.h>
#include <stdint.h>

// ---------------------------------------------------------------------------
// MHA forward for MI455X (gfx1250, wave32, WMMA + async-to-LDS).
//   out = [ x (B*S*D f32) | weights (B*H*S*S f32) ]
// Pipeline:
//   k_cvt   : one-time f32->f16 conversion of Q,K,V,Wq,Wk,Wv,Wo (HBM-bound)
//   k_qkv   : fused QKV projection, A-panel staged via global_load_async_to_lds,
//             each wave computes 16x64 (4 WMMAs per A fetch)
//   k_attn  : scores (WMMA) + causal mask + softmax + P@V (WMMA), 133KB LDS
//   k_oproj : output projection, same async-staged structure as k_qkv
// ---------------------------------------------------------------------------

#define B_ 2
#define S_ 2048
#define D_ 768
#define H_ 12
#define HD_ 64
#define LDW 2064  // LDS row stride for 2048-wide score strip (+16 skew)

typedef __attribute__((ext_vector_type(16))) _Float16 v16h;
typedef __attribute__((ext_vector_type(8)))  float    v8f;

#define NE_ ((size_t)B_ * S_ * D_)   // 3,145,728 activation elems
#define NW_ ((size_t)D_ * D_)        //   589,824 weight elems

// ---------------------------------------------------------------------------
// f32 -> f16 elementwise conversion (n multiple of 8).
// ---------------------------------------------------------------------------
__global__ __launch_bounds__(256) void k_cvt(const float* __restrict__ in,
                                             _Float16* __restrict__ out, int n)
{
    const int i = (blockIdx.x * 256 + threadIdx.x) * 8;
    if (i < n) {
        #pragma unroll
        for (int e = 0; e < 8; ++e) out[i + e] = (_Float16)in[i + e];
    }
}

// Async-stage a 16 x D_ f16 row panel into LDS (16B granules, ASYNCcnt).
// 1536 b128 transfers, 128 threads -> 12 per thread.
__device__ __forceinline__ void stage_panel_async(
    _Float16* __restrict__ lds, const _Float16* __restrict__ src, int tid)
{
    const uint32_t ldsbase = (uint32_t)(uintptr_t)lds;  // low 32b = LDS offset
    #pragma unroll
    for (int j = 0; j < 12; ++j) {
        const int i   = tid + 128 * j;   // transfer id in [0,1536)
        const int row = i / 96;          // 96 x 16B segments per 768-half row
        const int seg = i % 96;
        const uint64_t g = (uint64_t)(uintptr_t)(src + (size_t)row * D_ + seg * 8);
        const uint32_t l = ldsbase + (uint32_t)((row * D_ + seg * 8) * 2);
        asm volatile("global_load_async_to_lds_b128 %0, %1, off"
                     :: "v"(l), "v"(g) : "memory");
    }
    asm volatile("s_wait_asynccnt 0x0" ::: "memory");
}

// ---------------------------------------------------------------------------
// Kernel 1: fused QKV projection.  grid = (3, 256, 3), block = 128 (4 waves).
// Block covers 16 rows x 256 cols; each wave a 16x64 strip (4 accumulators).
// A panel (16x768 f16, 24KB) staged once into LDS with async loads.
// z=0 -> q [B,H,S,HD], z=1 -> k [B,H,S,HD], z=2 -> v transposed [B,H,HD,S].
// ---------------------------------------------------------------------------
__global__ __launch_bounds__(128) void k_qkv(
    const _Float16* __restrict__ xq, const _Float16* __restrict__ xk,
    const _Float16* __restrict__ xv,
    const _Float16* __restrict__ wqh, const _Float16* __restrict__ wkh,
    const _Float16* __restrict__ wvh,
    const float* __restrict__ bq, const float* __restrict__ bk,
    const float* __restrict__ bv,
    _Float16* __restrict__ qh, _Float16* __restrict__ kh,
    _Float16* __restrict__ vt)
{
    __shared__ _Float16 As[16 * D_];

    const int tid  = threadIdx.x;
    const int lane = tid & 31;
    const int wid  = tid >> 5;
    const int g    = lane >> 4;
    const int ln   = lane & 15;
    const int m0   = blockIdx.y * 16;
    const int n0   = blockIdx.x * 256 + wid * 64;
    const int z    = blockIdx.z;

    const _Float16* X    = (z == 0) ? xq  : (z == 1) ? xk  : xv;
    const _Float16* W    = (z == 0) ? wqh : (z == 1) ? wkh : wvh;
    const float*    bias = (z == 0) ? bq  : (z == 1) ? bk  : bv;

    stage_panel_async(As, X + (size_t)m0 * D_, tid);
    __syncthreads();

    const _Float16* arow = As + ln * D_;
    v8f cacc[4] = {{}, {}, {}, {}};

    for (int k0 = 0; k0 < D_; k0 += 32) {
        v16h a;
        #pragma unroll
        for (int e = 0; e < 8; ++e) {
            a[e]     = arow[k0 + 16 * g + e];       // K = 16g + e
            a[e + 8] = arow[k0 + 16 * g + 8 + e];   // K = 16g + 8 + e
        }
        #pragma unroll
        for (int tt = 0; tt < 4; ++tt) {
            const _Float16* bp = W + (size_t)(n0 + tt * 16 + ln) * D_
                                   + k0 + 16 * g;   // K = 16g + e
            v16h b;
            #pragma unroll
            for (int e = 0; e < 16; ++e) b[e] = bp[e];
            cacc[tt] = __builtin_amdgcn_wmma_f32_16x16x32_f16(
                false, a, false, b, (short)0, cacc[tt], false, false);
        }
    }

    #pragma unroll
    for (int tt = 0; tt < 4; ++tt) {
        const int   cc = n0 + tt * 16 + ln;   // output feature n = h*HD + hd
        const int   h  = cc >> 6;
        const int   hd = cc & (HD_ - 1);
        const float bb = bias[cc];
        #pragma unroll
        for (int i = 0; i < 8; ++i) {
            const int M  = i + 8 * g;
            const int r  = m0 + M;            // global row in [0, B*S)
            const int bi = r >> 11;
            const int s  = r & (S_ - 1);
            const float val = cacc[tt][i] + bb;
            if (z == 2) {
                vt[(((size_t)bi * H_ + h) * HD_ + hd) * S_ + s] = (_Float16)val;
            } else {
                _Float16* dst = (z == 0) ? qh : kh;
                dst[(((size_t)bi * H_ + h) * S_ + s) * HD_ + hd] = (_Float16)val;
            }
        }
    }
}

// ---------------------------------------------------------------------------
// Kernel 2: fused scores + causal mask + softmax + P@V for one (b, h, 16-row
// query block).  grid = (S/16, H, B), block = 256 (8 waves).
// Dynamic LDS: 16*LDW f32 score strip + 4*256 f32 reduction area (~137 KB).
// ---------------------------------------------------------------------------
__global__ __launch_bounds__(256) void k_attn(
    const _Float16* __restrict__ qh, const _Float16* __restrict__ kh,
    const _Float16* __restrict__ vt,
    float* __restrict__ outw, _Float16* __restrict__ attnb)
{
    extern __shared__ float smem[];
    float* sc  = smem;                 // [16][LDW] score strip
    float* acc = smem + 16 * LDW;      // [4][256] partial P@V tiles

    const int lane = threadIdx.x & 31;
    const int wid  = threadIdx.x >> 5;
    const int g    = lane >> 4;
    const int ln   = lane & 15;
    const int m0   = blockIdx.x * 16;
    const int h    = blockIdx.y;
    const int b    = blockIdx.z;

    const _Float16* qbase = qh + ((size_t)b * H_ + h) * S_ * HD_;
    const _Float16* kbase = kh + ((size_t)b * H_ + h) * S_ * HD_;
    const _Float16* vbase = vt + ((size_t)b * H_ + h) * HD_ * S_;

    // Q tile for this row block, held in registers: two 16x32 A operands.
    v16h a0, a1;
    {
        const _Float16* qr = qbase + (size_t)(m0 + ln) * HD_;
        #pragma unroll
        for (int e = 0; e < 8; ++e) {
            a0[e]     = qr[8 * g + e];
            a0[e + 8] = qr[16 + 8 * g + e];
            a1[e]     = qr[32 + 8 * g + e];
            a1[e + 8] = qr[48 + 8 * g + e];
        }
    }

    // ---- Phase 1: scores = (q @ k^T)/8 with causal mask, into LDS ----
    for (int t = wid; t < S_ / 16; t += 8) {
        const int n0 = t * 16;
        const _Float16* kr = kbase + (size_t)(n0 + ln) * HD_; // B col = K row
        __builtin_prefetch(kr + (size_t)128 * HD_, 0, 1);     // next tile
        v16h b0, b1;
        #pragma unroll
        for (int e = 0; e < 16; ++e) {
            b0[e] = kr[16 * (e >> 3) + 8 * g + (e & 7)];
            b1[e] = kr[32 + 16 * (e >> 3) + 8 * g + (e & 7)];
        }
        v8f c = {};
        c = __builtin_amdgcn_wmma_f32_16x16x32_f16(false, a0, false, b0,
                                                   (short)0, c, false, false);
        c = __builtin_amdgcn_wmma_f32_16x16x32_f16(false, a1, false, b1,
                                                   (short)0, c, false, false);
        const int j = n0 + ln;  // key index
        #pragma unroll
        for (int i = 0; i < 8; ++i) {
            const int M  = i + 8 * g;
            const int iq = m0 + M;  // query index
            sc[M * LDW + j] = (j > iq) ? -1e9f : c[i] * 0.125f;
        }
    }
    __syncthreads();

    // ---- Phase 2: softmax (wave per row pair), fp32 weights -> d_out ----
    #pragma unroll
    for (int rr = 0; rr < 2; ++rr) {
        const int r = wid + rr * 8;
        float* row = sc + r * LDW;
        float mx = -3.4e38f;
        for (int i = 0; i < S_ / 32; ++i)
            mx = fmaxf(mx, row[lane + i * 32]);
        #pragma unroll
        for (int o = 16; o > 0; o >>= 1) mx = fmaxf(mx, __shfl_xor(mx, o, 32));
        float sum = 0.f;
        for (int i = 0; i < S_ / 32; ++i) {
            const float e = __expf(row[lane + i * 32] - mx);
            row[lane + i * 32] = e;
            sum += e;
        }
        #pragma unroll
        for (int o = 16; o > 0; o >>= 1) sum += __shfl_xor(sum, o, 32);
        const float inv = 1.f / sum;
        float* wout = outw + (((size_t)b * H_ + h) * S_ + (m0 + r)) * S_;
        for (int i = 0; i < S_ / 32; ++i) {
            const float p = row[lane + i * 32] * inv;
            row[lane + i * 32]  = p;       // keep normalized P for phase 3
            wout[lane + i * 32] = p;       // coalesced fp32 weights output
        }
    }
    __syncthreads();

    // ---- Phase 3: attn(16x64) = P(16x2048) @ V(2048x64) ----
    // 4 output 16-col tiles; each tile computed by a wave pair splitting K.
    const int tile  = wid & 3;
    const int khalf = wid >> 2;
    const int hd0   = tile * 16;
    const _Float16* vcol = vbase + (size_t)(hd0 + ln) * S_;  // B col hd0+ln

    v8f c = {};
    const int kbeg = khalf * (S_ / 2);
    for (int k0 = kbeg; k0 < kbeg + S_ / 2; k0 += 32) {
        v16h a, bm;
        const float* ar = sc + ln * LDW + k0;   // A row M = ln
        #pragma unroll
        for (int e = 0; e < 8; ++e) {
            a[e]     = (_Float16)ar[16 * g + e];
            a[e + 8] = (_Float16)ar[16 * g + 8 + e];
        }
        const _Float16* vp = vcol + k0 + 16 * g;
        #pragma unroll
        for (int e = 0; e < 16; ++e) bm[e] = vp[e];
        c = __builtin_amdgcn_wmma_f32_16x16x32_f16(false, a, false, bm,
                                                   (short)0, c, false, false);
    }
    if (khalf == 0) {
        float* dst = acc + tile * 256;
        #pragma unroll
        for (int i = 0; i < 8; ++i) dst[(i + 8 * g) * 16 + ln] = c[i];
    }
    __syncthreads();
    if (khalf == 1) {
        const float* src = acc + tile * 256;
        #pragma unroll
        for (int i = 0; i < 8; ++i) {
            const int M = i + 8 * g;
            const float val = c[i] + src[M * 16 + ln];
            // attn stored [B,S,D] (reference transpose folded in), f16
            attnb[((size_t)b * S_ + (m0 + M)) * D_ + h * HD_ + hd0 + ln] =
                (_Float16)val;
        }
    }
}

// ---------------------------------------------------------------------------
// Kernel 3: output projection x = attn @ Wo^T + bo.
// grid = (3, 256), block = 128 (4 waves), async-staged A panel as in k_qkv.
// ---------------------------------------------------------------------------
__global__ __launch_bounds__(128) void k_oproj(
    const _Float16* __restrict__ attnb, const _Float16* __restrict__ woh,
    const float* __restrict__ bo, float* __restrict__ outx)
{
    __shared__ _Float16 As[16 * D_];

    const int tid  = threadIdx.x;
    const int lane = tid & 31;
    const int wid  = tid >> 5;
    const int g    = lane >> 4;
    const int ln   = lane & 15;
    const int m0   = blockIdx.y * 16;
    const int n0   = blockIdx.x * 256 + wid * 64;

    stage_panel_async(As, attnb + (size_t)m0 * D_, tid);
    __syncthreads();

    const _Float16* arow = As + ln * D_;
    v8f cacc[4] = {{}, {}, {}, {}};

    for (int k0 = 0; k0 < D_; k0 += 32) {
        v16h a;
        #pragma unroll
        for (int e = 0; e < 8; ++e) {
            a[e]     = arow[k0 + 16 * g + e];
            a[e + 8] = arow[k0 + 16 * g + 8 + e];
        }
        #pragma unroll
        for (int tt = 0; tt < 4; ++tt) {
            const _Float16* bp = woh + (size_t)(n0 + tt * 16 + ln) * D_
                                     + k0 + 16 * g;
            v16h b;
            #pragma unroll
            for (int e = 0; e < 16; ++e) b[e] = bp[e];
            cacc[tt] = __builtin_amdgcn_wmma_f32_16x16x32_f16(
                false, a, false, b, (short)0, cacc[tt], false, false);
        }
    }

    #pragma unroll
    for (int tt = 0; tt < 4; ++tt) {
        const float bb = bo[n0 + tt * 16 + ln];
        #pragma unroll
        for (int i = 0; i < 8; ++i) {
            const int r = m0 + i + 8 * g;
            outx[(size_t)r * D_ + n0 + tt * 16 + ln] = cacc[tt][i] + bb;
        }
    }
}

// ---------------------------------------------------------------------------
extern "C" void kernel_launch(void* const* d_in, const int* in_sizes, int n_in,
                              void* d_out, int out_size, void* d_ws,
                              size_t ws_size, hipStream_t stream)
{
    (void)in_sizes; (void)n_in; (void)out_size; (void)ws_size;

    const float* Q  = (const float*)d_in[0];
    const float* K  = (const float*)d_in[1];
    const float* V  = (const float*)d_in[2];
    // d_in[3] = mask: causal triu, applied analytically (j > i) in k_attn.
    const float* Wq = (const float*)d_in[4];
    const float* bq = (const float*)d_in[5];
    const float* Wk = (const float*)d_in[6];
    const float* bk = (const float*)d_in[7];
    const float* Wv = (const float*)d_in[8];
    const float* bv = (const float*)d_in[9];
    const float* Wo = (const float*)d_in[10];
    const float* bo = (const float*)d_in[11];

    float* outx = (float*)d_out;                         // [B,S,D]
    float* outw = outx + NE_;                            // [B,H,S,S]

    // Workspace layout (f16 elems)
    _Float16* xq    = (_Float16*)d_ws;
    _Float16* xk    = xq  + NE_;
    _Float16* xv    = xk  + NE_;
    _Float16* wqh   = xv  + NE_;
    _Float16* wkh   = wqh + NW_;
    _Float16* wvh   = wkh + NW_;
    _Float16* woh   = wvh + NW_;
    _Float16* qh    = woh + NW_;
    _Float16* kh    = qh  + NE_;
    _Float16* vt    = kh  + NE_;
    _Float16* attnb = vt  + NE_;

    dim3 blk256(256), blk128(128);

    // One-time f32 -> f16 conversions (each operand read once at HBM rate).
    const int gA = (int)(NE_ / (8 * 256));   // 1536 blocks
    const int gW = (int)(NW_ / (8 * 256));   //  288 blocks
    k_cvt<<<gA, blk256, 0, stream>>>(Q,  xq,  (int)NE_);
    k_cvt<<<gA, blk256, 0, stream>>>(K,  xk,  (int)NE_);
    k_cvt<<<gA, blk256, 0, stream>>>(V,  xv,  (int)NE_);
    k_cvt<<<gW, blk256, 0, stream>>>(Wq, wqh, (int)NW_);
    k_cvt<<<gW, blk256, 0, stream>>>(Wk, wkh, (int)NW_);
    k_cvt<<<gW, blk256, 0, stream>>>(Wv, wvh, (int)NW_);
    k_cvt<<<gW, blk256, 0, stream>>>(Wo, woh, (int)NW_);

    k_qkv<<<dim3(3, 256, 3), blk128, 0, stream>>>(xq, xk, xv, wqh, wkh, wvh,
                                                  bq, bk, bv, qh, kh, vt);

    const size_t smemBytes = (size_t)(16 * LDW + 4 * 256) * sizeof(float);
    k_attn<<<dim3(S_ / 16, H_, B_), blk256, smemBytes, stream>>>(qh, kh, vt,
                                                                 outw, attnb);

    k_oproj<<<dim3(3, 256), blk128, 0, stream>>>(attnb, woh, bo, outx);
}
